// PositionAttention_20925080666259
// MI455X (gfx1250) — matrix operand order repaired and checked by reading, not compiled, not run on hardware
//
#include <hip/hip_runtime.h>

// ---------------------------------------------------------------------------
// PositionAttention for MI455X (gfx1250), bf16 WMMA pipeline with async-LDS
// double buffering.
//   N=4, C=2048, H=W=64 -> HW=4096, CR=256
//   xbT = bf16(x)^T per batch            (HW x C)
//   Q   = (W1 X)^T + b1                  (HW x CR)  bf16   [EPI1]
//   Kt  = (W2 X)^T + b2                  (HW x CR)  bf16   [EPI1]
//   V   =  W3 X   + b3                   (C  x HW)  bf16   [EPI0]
//   S   =  Q Kt^T                        (HW x HW)  bf16, softmax rows -> P
//   out =  alpha * (V P^T) + x           f32        [EPI3]
// Every GEMM consumes B in N x K layout so all staging is 16B-contiguous and
// rides GLOBAL_LOAD_ASYNC_TO_LDS_B128 (ASYNCcnt) when available.
// ---------------------------------------------------------------------------

typedef __attribute__((ext_vector_type(16))) __bf16 bf16x16;
typedef __attribute__((ext_vector_type(8)))  __bf16 bf16x8;
typedef __attribute__((ext_vector_type(8)))  float  f32x8;
typedef int v4i __attribute__((vector_size(16)));

union V16 { bf16x16 v; bf16x8 h[2]; };

constexpr int NB = 4;
constexpr int CC = 2048;
constexpr int HW = 4096;
constexpr int CR = 256;

// 256 threads = 8 waves in a 2(m) x 4(n) grid; each wave owns a 32x64 tile.
constexpr int BM  = 64;
constexpr int BN  = 256;
constexpr int BK  = 32;
constexpr int BKP = BK + 8;   // padded LDS stride: 80B rows, 16B-aligned, conflict-free

#if defined(__gfx1250__) && __has_builtin(__builtin_amdgcn_global_load_async_to_lds_b128)
#define USE_ASYNC_LDS 1
#else
#define USE_ASYNC_LDS 0
#endif

__device__ __forceinline__ void cp16_to_lds(const __bf16* g, __bf16* l) {
#if USE_ASYNC_LDS
  __builtin_amdgcn_global_load_async_to_lds_b128(
      (__attribute__((address_space(1))) v4i*)g,
      (__attribute__((address_space(3))) v4i*)l, 0, 0);
#else
  *(uint4*)l = *(const uint4*)g;
#endif
}

__device__ __forceinline__ void wait_async_pending() {   // allow next tile in flight
#if USE_ASYNC_LDS
#if __has_builtin(__builtin_amdgcn_s_wait_asynccnt)
  __builtin_amdgcn_s_wait_asynccnt(5);
#else
  asm volatile("s_wait_asynccnt 0x5" ::: "memory");
#endif
#endif
}
__device__ __forceinline__ void wait_async_zero() {
#if USE_ASYNC_LDS
#if __has_builtin(__builtin_amdgcn_s_wait_asynccnt)
  __builtin_amdgcn_s_wait_asynccnt(0);
#else
  asm volatile("s_wait_asynccnt 0x0" ::: "memory");
#endif
#endif
}

// ---------------------------------------------------------------------------
__global__ void f32_to_bf16_kernel(const float* __restrict__ in,
                                   __bf16* __restrict__ out, long long n) {
  long long i = (long long)blockIdx.x * blockDim.x + threadIdx.x;
  if (i < n) out[i] = (__bf16)in[i];
}

// Per batch: in (C x HW) f32 -> out (HW x C) bf16, 32x32 LDS-tiled transpose.
__global__ __launch_bounds__(256)
void f32_to_bf16_T_kernel(const float* __restrict__ in, __bf16* __restrict__ out) {
  __shared__ float t[32][33];
  const int tx = threadIdx.x & 31, ty = threadIdx.x >> 5;   // 32 x 8
  const int p0 = blockIdx.x * 32, c0 = blockIdx.y * 32, b = blockIdx.z;
  const float* ib = in + (size_t)b * CC * HW;
  __bf16* ob = out + (size_t)b * HW * CC;
#pragma unroll
  for (int j = ty; j < 32; j += 8)
    t[j][tx] = ib[(size_t)(c0 + j) * HW + p0 + tx];
  __syncthreads();
#pragma unroll
  for (int j = ty; j < 32; j += 8)
    ob[(size_t)(p0 + j) * CC + c0 + tx] = (__bf16)t[tx][j];
}

// ---------------------------------------------------------------------------
// Tiled bf16 GEMM: D = A(MxK) * Bsrc, f32 accumulate, B given N x K
// (Bsrc[n*ldb + k]).  Double-buffered LDS, async global->LDS staging.
// EPI: 0 = bf16 store (+optional bias[m])
//      1 = bf16 store transposed, +bias[m]   (D[n*ldd + m])
//      3 = f32 store: alpha*acc + X
template<int EPI>
__global__ __launch_bounds__(256)
void gemm_bf16_kernel(const __bf16* __restrict__ A, long long sAb, int lda,
                      const __bf16* __restrict__ B, long long sBb, int ldb,
                      int Kdim,
                      const float* __restrict__ bias,
                      void* __restrict__ Dv, long long sDb, int ldd,
                      const float* __restrict__ alphap,
                      const float* __restrict__ X, long long sXb)
{
  alignas(16) __shared__ __bf16 sA[2][BM * BKP];   //  2 x  5.0 KB
  alignas(16) __shared__ __bf16 sB[2][BN * BKP];   //  2 x 20.0 KB

  const int b    = blockIdx.z;
  const int m0   = blockIdx.y * BM;
  const int n0   = blockIdx.x * BN;
  const int tid  = threadIdx.x;
  const int lane = tid & 31;
  const int wid  = tid >> 5;
  const int wm   = (wid & 1) * 32;    // wave m-offset
  const int wn   = (wid >> 1) * 64;   // wave n-offset

  const __bf16* Ab = A + (size_t)b * sAb;
  const __bf16* Bb = B + (size_t)b * sBb;

  f32x8 acc[2][4] = {};

  // Staging shares: A 64x32 = 256 16B-chunks (1/thread); B 256x32 = 1024 (4/thread).
  const int a_m = tid >> 2;            // 0..63
  const int a_k = (tid & 3) * 8;       // 0,8,16,24

  const int row16 = lane & 15;
  const int hi    = lane >> 4;         // 0/1 half-wave

  auto stage = [&](int k0, int bufi) {
    cp16_to_lds(&Ab[(size_t)(m0 + a_m) * lda + k0 + a_k],
                &sA[bufi][a_m * BKP + a_k]);
#pragma unroll
    for (int c = 0; c < 4; ++c) {
      const int q  = tid + 256 * c;    // 0..1023
      const int n  = q >> 2;           // 0..255
      const int kc = (q & 3) * 8;
      cp16_to_lds(&Bb[(size_t)(n0 + n) * ldb + k0 + kc],
                  &sB[bufi][n * BKP + kc]);
    }
  };

  const int iters = Kdim / BK;
  stage(0, 0);
  for (int i = 0; i < iters; ++i) {
    const int cur = i & 1;
    const bool more = (i + 1) < iters;
    if (more) stage(i + 1, cur ^ 1);
    if (more) wait_async_pending(); else wait_async_zero();
    __syncthreads();

    // A 16x32 bf16 frag: lane m = lane%16; lo-half k={0..7,16..23}, hi-half +8.
    V16 afrag[2];
#pragma unroll
    for (int mt = 0; mt < 2; ++mt) {
      const __bf16* base = &sA[cur][(wm + mt * 16 + row16) * BKP + hi * 8];
      afrag[mt].h[0] = *(const bf16x8*)(base);
      afrag[mt].h[1] = *(const bf16x8*)(base + 16);
    }
    // B 32x16 bf16 frag: lane n = lane%16; lo-half k={0..7,16..23}, hi-half +8.
    V16 bfrag[4];
#pragma unroll
    for (int nt = 0; nt < 4; ++nt) {
      const __bf16* base = &sB[cur][(wn + nt * 16 + row16) * BKP + hi * 8];
      bfrag[nt].h[0] = *(const bf16x8*)(base);
      bfrag[nt].h[1] = *(const bf16x8*)(base + 16);
    }
#pragma unroll
    for (int mt = 0; mt < 2; ++mt)
#pragma unroll
      for (int nt = 0; nt < 4; ++nt)
        acc[mt][nt] = __builtin_amdgcn_wmma_f32_16x16x32_bf16(
            false, afrag[mt].v, false, bfrag[nt].v,
            (short)0, acc[mt][nt], false, false);

    __syncthreads();   // all waves done reading buf[cur] before it is restaged
  }

  // C/D layout: VGPR r -> M = r + 8*hi, N = lane%16.
  const int mhi = hi * 8;
#pragma unroll
  for (int mt = 0; mt < 2; ++mt) {
#pragma unroll
    for (int nt = 0; nt < 4; ++nt) {
      const int m_base = m0 + wm + mt * 16 + mhi;
      const int n_g    = n0 + wn + nt * 16 + row16;
#pragma unroll
      for (int r = 0; r < 8; ++r) {
        const int m_g = m_base + r;
        float v = acc[mt][nt][r];
        if constexpr (EPI == 0) {
          if (bias) v += bias[m_g];
          ((__bf16*)Dv)[(size_t)b * sDb + (size_t)m_g * ldd + n_g] = (__bf16)v;
        } else if constexpr (EPI == 1) {
          v += bias[m_g];
          ((__bf16*)Dv)[(size_t)b * sDb + (size_t)n_g * ldd + m_g] = (__bf16)v;
        } else {  // EPI == 3
          const size_t o = (size_t)b * sDb + (size_t)m_g * ldd + n_g;
          ((float*)Dv)[o] = alphap[0] * v + X[(size_t)b * sXb + (size_t)m_g * ldd + n_g];
        }
      }
    }
  }
}

// ---------------------------------------------------------------------------
// Row softmax, in place on bf16 scores. One block (256 thr) per row of 4096.
__global__ __launch_bounds__(256)
void softmax_rows_kernel(__bf16* __restrict__ S) {
  __shared__ float red[256];
  const size_t row = blockIdx.x;
  __bf16* p = S + row * (size_t)HW;
  const int t = threadIdx.x;

  float v[16];
  float m = -1e30f;
#pragma unroll
  for (int i = 0; i < 16; ++i) { v[i] = (float)p[t + (i << 8)]; m = fmaxf(m, v[i]); }
  red[t] = m; __syncthreads();
  for (int s = 128; s > 0; s >>= 1) {
    if (t < s) red[t] = fmaxf(red[t], red[t + s]);
    __syncthreads();
  }
  m = red[0]; __syncthreads();

  float sum = 0.f;
#pragma unroll
  for (int i = 0; i < 16; ++i) { v[i] = __expf(v[i] - m); sum += v[i]; }
  red[t] = sum; __syncthreads();
  for (int s = 128; s > 0; s >>= 1) {
    if (t < s) red[t] += red[t + s];
    __syncthreads();
  }
  const float inv = 1.f / red[0];
#pragma unroll
  for (int i = 0; i < 16; ++i) p[t + (i << 8)] = (__bf16)(v[i] * inv);
}

// ---------------------------------------------------------------------------
extern "C" void kernel_launch(void* const* d_in, const int* in_sizes, int n_in,
                              void* d_out, int out_size, void* d_ws, size_t ws_size,
                              hipStream_t stream) {
  const float* x  = (const float*)d_in[0];
  const float* w1 = (const float*)d_in[1];
  const float* b1 = (const float*)d_in[2];
  const float* w2 = (const float*)d_in[3];
  const float* b2 = (const float*)d_in[4];
  const float* w3 = (const float*)d_in[5];
  const float* b3 = (const float*)d_in[6];
  const float* al = (const float*)d_in[7];
  float* out = (float*)d_out;

  constexpr long long nXB = (long long)NB * CC * HW;   // 33,554,432
  constexpr long long nW1 = (long long)CR * CC;        // 524,288
  constexpr long long nW3 = (long long)CC * CC;        // 4,194,304
  constexpr long long nQ  = (long long)NB * HW * CR;   // 4,194,304
  constexpr long long nS  = (long long)NB * HW * HW;   // 67,108,864

  char* ws = (char*)d_ws;
  size_t off = 0;
  __bf16* xbT = (__bf16*)(ws + off); off += (size_t)nXB * 2;  // (HW x C) per batch
  __bf16* wq  = (__bf16*)(ws + off); off += (size_t)nW1 * 2;
  __bf16* wk  = (__bf16*)(ws + off); off += (size_t)nW1 * 2;
  __bf16* wv  = (__bf16*)(ws + off); off += (size_t)nW3 * 2;
  __bf16* Qb  = (__bf16*)(ws + off); off += (size_t)nQ  * 2;  // (HW x CR)
  __bf16* Kt  = (__bf16*)(ws + off); off += (size_t)nQ  * 2;  // (HW x CR)
  __bf16* Vb  = (__bf16*)(ws + off); off += (size_t)nXB * 2;  // (C x HW)
  __bf16* Sb  = (__bf16*)(ws + off); off += (size_t)nS  * 2;  // (HW x HW)

  // ---- 0) convert (and transpose x) to bf16 ----
  f32_to_bf16_T_kernel<<<dim3(HW / 32, CC / 32, NB), 256, 0, stream>>>(x, xbT);
  f32_to_bf16_kernel<<<(unsigned)((nW1 + 255) / 256), 256, 0, stream>>>(w1, wq, nW1);
  f32_to_bf16_kernel<<<(unsigned)((nW1 + 255) / 256), 256, 0, stream>>>(w2, wk, nW1);
  f32_to_bf16_kernel<<<(unsigned)((nW3 + 255) / 256), 256, 0, stream>>>(w3, wv, nW3);

  const long long sX  = (long long)CC * HW;   // per-batch x / v / out stride
  const long long sQK = (long long)HW * CR;   // per-batch q / k stride
  const long long sS  = (long long)HW * HW;   // per-batch score stride

  // ---- 1) Q = (W1 X)^T + b1 : (HW x CR).  B = xbT in N x K form. ----
  gemm_bf16_kernel<1><<<dim3(HW / BN, CR / BM, NB), 256, 0, stream>>>(
      wq, 0, CC, xbT, sX, CC, CC, b1, Qb, sQK, CR, nullptr, nullptr, 0);
  // ---- 2) Kt = (W2 X)^T + b2 : (HW x CR) ----
  gemm_bf16_kernel<1><<<dim3(HW / BN, CR / BM, NB), 256, 0, stream>>>(
      wk, 0, CC, xbT, sX, CC, CC, b2, Kt, sQK, CR, nullptr, nullptr, 0);
  // ---- 3) V = W3 X + b3 : (C x HW) ----
  gemm_bf16_kernel<0><<<dim3(HW / BN, CC / BM, NB), 256, 0, stream>>>(
      wv, 0, CC, xbT, sX, CC, CC, b3, Vb, sX, HW, nullptr, nullptr, 0);
  // ---- 4) S = Q Kt^T : (HW x HW), K-dim = CR; B = Kt already N x K ----
  gemm_bf16_kernel<0><<<dim3(HW / BN, HW / BM, NB), 256, 0, stream>>>(
      Qb, sQK, CR, Kt, sQK, CR, CR, nullptr, Sb, sS, HW, nullptr, nullptr, 0);
  // ---- 5) P = softmax rows of S (in place) ----
  softmax_rows_kernel<<<NB * HW, 256, 0, stream>>>(Sb);
  // ---- 6) out = alpha * (V P^T) + x ; B = P (N=i, K=j) ----
  gemm_bf16_kernel<3><<<dim3(HW / BN, CC / BM, NB), 256, 0, stream>>>(
      Vb, sX, HW, Sb, sS, HW, HW, nullptr, out, sX, HW, al, x, sX);

  (void)in_sizes; (void)n_in; (void)out_size; (void)ws_size;
}
